// EdgeCondConv_40029095199352
// MI455X (gfx1250) — matrix operand order repaired and checked by reading, not compiled, not run on hardware
//
#include <hip/hip_runtime.h>
#include <hip/hip_bf16.h>
#include <stdint.h>

typedef __attribute__((ext_vector_type(16))) __bf16 v16bf;
typedef __attribute__((ext_vector_type(8)))  __bf16 v8bf;
typedef __attribute__((ext_vector_type(4)))  __bf16 v4bf;
typedef __attribute__((ext_vector_type(8)))  float  v8f;

#define NN      100000
#define EE      1600000
#define INDIM   128
#define EDIM    16
#define HIDD    128
#define MSGIN   146
#define KPAD    160                     // 146 padded to a multiple of 32 (5 K-chunks)
#define LNEPS   1e-5f
#define W1ELEMS (KPAD * HIDD)           // 20480 bf16 (40 KB)
#define W2ELEMS (HIDD * HIDD)           // 16384 bf16 (32 KB)
#define WBYTES  ((W1ELEMS + W2ELEMS) * 2)  // 73728 B, contiguous in workspace

__device__ __forceinline__ __bf16 tobf(float x) { return (__bf16)x; }

// ---------------------------------------------------------------------------
// Prep: zero scatter accumulator; convert weights to bf16 (W1 zero-padded
// 146 -> 160 K-rows). W1bf and W2bf are contiguous so the edge kernel can
// async-DMA them into LDS as one block.
// ---------------------------------------------------------------------------
__global__ void prep_kernel(const float* __restrict__ W1,
                            const float* __restrict__ W2,
                            const float* __restrict__ Ws,
                            __bf16* __restrict__ W1bf,
                            __bf16* __restrict__ W2bf,
                            __bf16* __restrict__ Wsbf,
                            float* __restrict__ msum)
{
    const long tid    = (long)blockIdx.x * blockDim.x + threadIdx.x;
    const long stride = (long)gridDim.x * blockDim.x;

    const long nm = (long)NN * HIDD;
    for (long j = tid; j < nm; j += stride) msum[j] = 0.0f;

    const long n1 = (long)KPAD * HIDD;
    for (long j = tid; j < n1; j += stride) {
        const int r = (int)(j / HIDD), c = (int)(j % HIDD);
        W1bf[j] = (r < MSGIN) ? tobf(W1[r * HIDD + c]) : tobf(0.0f);
    }

    const long n2 = (long)HIDD * HIDD;
    for (long j = tid; j < n2; j += stride) {
        W2bf[j] = tobf(W2[j]);
        Wsbf[j] = tobf(Ws[j]);
    }
}

// ---------------------------------------------------------------------------
// Edge MLP: one wave32 per 32-edge tile (two 16-row A blocks share every B
// fragment -> 2 WMMAs per B read). Weights live in LDS, staged once per block
// via the async global->LDS DMA path (ASYNCcnt).
//
// A-fragment (16-bit, 16x32): row M = lane&15, g = lane>>4;
//   elem j<8 -> K = g*8+j ; elem j>=8 -> K = 16+g*8+(j-8)
// B-fragment (32x16): column N = lane&15; elem j -> same K as A elem j.
// C/D: elem r -> row M = 8*g+r, column N = lane&15.
// ---------------------------------------------------------------------------
__global__ __launch_bounds__(256) void edge_mlp_kernel(
    const float* __restrict__ h,    const float* __restrict__ efeat,
    const int*   __restrict__ src,  const int*   __restrict__ dst,
    const int*   __restrict__ etype,
    const __bf16* __restrict__ W1bf,   // start of contiguous [W1bf|W2bf] block
    const float* __restrict__ b1,  const float* __restrict__ b2,
    float* __restrict__ msum)
{
    __shared__ __bf16 Wlds[W1ELEMS + W2ELEMS];   // 72 KB, block-shared
    __shared__ __bf16 Zs[8][32 * KPAD];          // 80 KB, per-wave staging
    const int wave = threadIdx.x >> 5;
    const int lane = threadIdx.x & 31;
    const int g    = lane >> 4;
    const int m    = lane & 15;
    const int e0   = (blockIdx.x * 8 + wave) * 32;   // grid exact: EE/(32*8)
    __bf16* Z = Zs[wave];

    // ---- async-stage bf16 weights (one contiguous 73728B block) into LDS ----
    {
        const uint64_t gbase = (uint64_t)(uintptr_t)W1bf;
        const uint32_t lbase = (uint32_t)(uintptr_t)(&Wlds[0]);
        for (int i = threadIdx.x; i < WBYTES / 16; i += 256) {
            const uint32_t loff = lbase + (uint32_t)i * 16u;
            const uint64_t ga   = gbase + (uint64_t)i * 16u;
            asm volatile("global_load_async_to_lds_b128 %0, %1, off"
                         :: "v"(loff), "v"(ga) : "memory");
        }
    }

    // ---- stage z = [h[src] | e_feat | onehot | pad] as bf16 (vectorized) ----
    const int se = src[e0 + lane];
#pragma unroll 8
    for (int r = 0; r < 32; ++r) {
        const int sr = __shfl(se, r, 32);
        const float4 hv = *(const float4*)&h[sr * INDIM + lane * 4];
        v4bf o = { tobf(hv.x), tobf(hv.y), tobf(hv.z), tobf(hv.w) };
        *(v4bf*)&Z[r * KPAD + lane * 4] = o;
    }
#pragma unroll
    for (int it = 0; it < 4; ++it) {
        const int r = it * 8 + (lane >> 2);
        const int c = (lane & 3) * 4;
        const float4 ev = *(const float4*)&efeat[(e0 + r) * EDIM + c];
        v4bf o = { tobf(ev.x), tobf(ev.y), tobf(ev.z), tobf(ev.w) };
        *(v4bf*)&Z[r * KPAD + INDIM + c] = o;
    }
    {
        const int t = etype[e0 + lane];
        v8bf o = {};
        o[0] = tobf(t == 0 ? 1.0f : 0.0f);
        o[1] = tobf(t == 1 ? 1.0f : 0.0f);
        *(v8bf*)&Z[lane * KPAD + 144] = o;       // cols 144..151 (onehot + pad)
        v8bf zz = {};
        *(v8bf*)&Z[lane * KPAD + 152] = zz;      // cols 152..159 (pad)
    }
    asm volatile("s_wait_asynccnt 0x0" ::: "memory");  // weight DMA done (this wave)
    __syncthreads();                                   // all waves: weights + z ready

    // ---- layer-1 A fragments: 2 row-blocks x 5 K-chunks ----
    v16bf A1[2][5];
#pragma unroll
    for (int s = 0; s < 2; ++s)
#pragma unroll
        for (int kc = 0; kc < 5; ++kc) {
            const __bf16* p = &Z[(s * 16 + m) * KPAD + kc * 32 + g * 8];
            v16bf a;
#pragma unroll
            for (int j = 0; j < 8; ++j) { a[j] = p[j]; a[j + 8] = p[j + 16]; }
            A1[s][kc] = a;
        }

    // ---- hidden = relu(z @ W1 + b1); stash to LDS per column tile ----
#pragma unroll
    for (int nt = 0; nt < 8; ++nt) {
        const float bias = b1[nt * 16 + m];
        v8f c0, c1;
#pragma unroll
        for (int r = 0; r < 8; ++r) { c0[r] = bias; c1[r] = bias; }
#pragma unroll
        for (int kc = 0; kc < 5; ++kc) {
            const __bf16* bp = &Wlds[(kc * 32 + g * 8) * HIDD + nt * 16 + m];
            v16bf b;
#pragma unroll
            for (int j = 0; j < 8; ++j) { b[j] = bp[j * HIDD]; b[j + 8] = bp[(j + 16) * HIDD]; }
            c0 = __builtin_amdgcn_wmma_f32_16x16x32_bf16(false, A1[0][kc], false, b,
                                                         (short)0, c0, false, false);
            c1 = __builtin_amdgcn_wmma_f32_16x16x32_bf16(false, A1[1][kc], false, b,
                                                         (short)0, c1, false, false);
        }
#pragma unroll
        for (int r = 0; r < 8; ++r) {
            Z[(g * 8 + r) * HIDD + nt * 16 + m]      = tobf(fmaxf(c0[r], 0.0f));
            Z[(16 + g * 8 + r) * HIDD + nt * 16 + m] = tobf(fmaxf(c1[r], 0.0f));
        }
    }
    __builtin_amdgcn_wave_barrier();   // Z reuse is wave-local; DS is in-order per wave

    // ---- layer-2 A fragments: 2 row-blocks x 4 K-chunks ----
    v16bf A2[2][4];
#pragma unroll
    for (int s = 0; s < 2; ++s)
#pragma unroll
        for (int kc = 0; kc < 4; ++kc) {
            const __bf16* p = &Z[(s * 16 + m) * HIDD + kc * 32 + g * 8];
            v16bf a;
#pragma unroll
            for (int j = 0; j < 8; ++j) { a[j] = p[j]; a[j + 8] = p[j + 16]; }
            A2[s][kc] = a;
        }

    // destination row offsets for the scatter (broadcast one gather per lane)
    const int de = dst[e0 + lane];
    int dr[16];
#pragma unroll
    for (int r = 0; r < 8; ++r) {
        dr[r]     = __shfl(de, g * 8 + r, 32) * HIDD;
        dr[8 + r] = __shfl(de, 16 + g * 8 + r, 32) * HIDD;
    }

    // ---- m = hidden @ W2 + b2, atomic scatter-add to msum[dst] ----
#pragma unroll
    for (int nt = 0; nt < 8; ++nt) {
        const int col = nt * 16 + m;
        const float bias = b2[col];
        v8f c0, c1;
#pragma unroll
        for (int r = 0; r < 8; ++r) { c0[r] = bias; c1[r] = bias; }
#pragma unroll
        for (int kc = 0; kc < 4; ++kc) {
            const __bf16* bp = &Wlds[W1ELEMS + (kc * 32 + g * 8) * HIDD + nt * 16 + m];
            v16bf b;
#pragma unroll
            for (int j = 0; j < 8; ++j) { b[j] = bp[j * HIDD]; b[j + 8] = bp[(j + 16) * HIDD]; }
            c0 = __builtin_amdgcn_wmma_f32_16x16x32_bf16(false, A2[0][kc], false, b,
                                                         (short)0, c0, false, false);
            c1 = __builtin_amdgcn_wmma_f32_16x16x32_bf16(false, A2[1][kc], false, b,
                                                         (short)0, c1, false, false);
        }
#pragma unroll
        for (int r = 0; r < 8; ++r) {
            unsafeAtomicAdd(&msum[dr[r] + col],     c0[r]);   // global_atomic_add_f32
            unsafeAtomicAdd(&msum[dr[8 + r] + col], c1[r]);
        }
    }
}

// ---------------------------------------------------------------------------
// Node update: out = LayerNorm(relu(h @ Ws + bs + m_sum))
// One wave32 per 16-node tile; tail waves clamp to tile 0 (identical redundant
// writes, no atomics) so block barriers stay uniform.
// ---------------------------------------------------------------------------
__global__ __launch_bounds__(256) void node_update_kernel(
    const float* __restrict__ h,    const __bf16* __restrict__ Wsbf,
    const float* __restrict__ bs,   const float* __restrict__ msum,
    const float* __restrict__ gamma,const float* __restrict__ beta,
    float* __restrict__ out)
{
    __shared__ __bf16 Hs[8][16 * INDIM];   // 32 KB
    __shared__ float  Os[8][16 * HIDD];    // 64 KB
    __shared__ float  Mu[8][16];
    __shared__ float  Rs[8][16];
    const int wave = threadIdx.x >> 5;
    const int lane = threadIdx.x & 31;
    const int g    = lane >> 4;
    const int m    = lane & 15;
    int tile = blockIdx.x * 8 + wave;
    if (tile >= NN / 16) tile = 0;
    const int n0 = tile * 16;

    __bf16* H = Hs[wave];
    float*  O = Os[wave];

    for (int i = lane; i < 16 * INDIM; i += 32) {
        const int r = i >> 7, c = i & 127;
        H[i] = tobf(h[(n0 + r) * INDIM + c]);
    }
    __syncthreads();

    v16bf A[4];
#pragma unroll
    for (int kc = 0; kc < 4; ++kc) {
        const __bf16* p = &H[m * INDIM + kc * 32 + g * 8];
        v16bf a;
#pragma unroll
        for (int j = 0; j < 8; ++j) { a[j] = p[j]; a[j + 8] = p[j + 16]; }
        A[kc] = a;
    }

#pragma unroll
    for (int nt = 0; nt < 8; ++nt) {
        const int col = nt * 16 + m;
        const float bias = bs[col];
        v8f c;
#pragma unroll
        for (int r = 0; r < 8; ++r) c[r] = bias;
#pragma unroll
        for (int kc = 0; kc < 4; ++kc) {
            const __bf16* bp = &Wsbf[(kc * 32 + g * 8) * HIDD + nt * 16 + m];
            v16bf b;
#pragma unroll
            for (int j = 0; j < 8; ++j) { b[j] = bp[j * HIDD]; b[j + 8] = bp[(j + 16) * HIDD]; }
            c = __builtin_amdgcn_wmma_f32_16x16x32_bf16(false, A[kc], false, b,
                                                        (short)0, c, false, false);
        }
#pragma unroll
        for (int r = 0; r < 8; ++r) {
            const int row = g * 8 + r;
            const float v = c[r] + msum[(n0 + row) * HIDD + col];
            O[row * HIDD + col] = fmaxf(v, 0.0f);
        }
    }
    __syncthreads();

    if (lane < 16) {
        float s = 0.0f, ss = 0.0f;
        for (int c = 0; c < HIDD; ++c) {
            const float v = O[lane * HIDD + c];
            s += v; ss += v * v;
        }
        const float mu  = s * (1.0f / HIDD);
        const float var = ss * (1.0f / HIDD) - mu * mu;
        Mu[wave][lane] = mu;
        Rs[wave][lane] = rsqrtf(var + LNEPS);
    }
    __syncthreads();

    for (int i = lane; i < 16 * HIDD; i += 32) {
        const int r = i >> 7, c = i & 127;
        out[(n0 + r) * HIDD + c] = (O[i] - Mu[wave][r]) * Rs[wave][r] * gamma[c] + beta[c];
    }
}

// ---------------------------------------------------------------------------
extern "C" void kernel_launch(void* const* d_in, const int* in_sizes, int n_in,
                              void* d_out, int out_size, void* d_ws, size_t ws_size,
                              hipStream_t stream)
{
    const float* h     = (const float*)d_in[0];
    const float* efeat = (const float*)d_in[1];
    const int*   src   = (const int*)  d_in[2];
    const int*   dst   = (const int*)  d_in[3];
    const int*   etype = (const int*)  d_in[4];
    const float* W1    = (const float*)d_in[5];
    const float* b1    = (const float*)d_in[6];
    const float* W2    = (const float*)d_in[7];
    const float* b2    = (const float*)d_in[8];
    const float* Ws    = (const float*)d_in[9];
    const float* bs    = (const float*)d_in[10];
    const float* gamma = (const float*)d_in[11];
    const float* beta  = (const float*)d_in[12];
    float* out = (float*)d_out;

    char* ws = (char*)d_ws;
    float*  msum = (float*)ws;   ws += (size_t)NN * HIDD * sizeof(float);   // 51.2 MB
    __bf16* W1bf = (__bf16*)ws;  ws += (size_t)W1ELEMS * sizeof(__bf16);    // contiguous with W2bf
    __bf16* W2bf = (__bf16*)ws;  ws += (size_t)W2ELEMS * sizeof(__bf16);
    __bf16* Wsbf = (__bf16*)ws;

    prep_kernel<<<2048, 256, 0, stream>>>(W1, W2, Ws, W1bf, W2bf, Wsbf, msum);

    // EE = 1,600,000 -> 50,000 32-edge tiles -> exactly 6,250 blocks of 8 waves
    edge_mlp_kernel<<<EE / (32 * 8), 256, 0, stream>>>(
        h, efeat, src, dst, etype, W1bf, b1, b2, msum);

    // NN = 100,000 -> 6,250 16-node tiles -> 782 blocks (tail clamps to tile 0)
    node_update_kernel<<<(NN / 16 + 7) / 8, 256, 0, stream>>>(
        h, Wsbf, bs, msum, gamma, beta, out);
}
